// NMT_68908455297601
// MI455X (gfx1250) — matrix operand order repaired and checked by reading, not compiled, not run on hardware
//
#include <hip/hip_runtime.h>
#include <hip/hip_bf16.h>
#include <stdint.h>
#include <stddef.h>

// ---------------- problem constants ----------------
#define S_LEN  50
#define T_LEN  50
#define BATCH  128
#define EMB    300
#define EMBP   320      // EMB padded to 32
#define EH     512      // encoder hidden
#define G_EH   2048     // 4*EH
#define DH     1024     // decoder hidden
#define G_DH   4096     // 4*DH
#define DINK   1324     // DH + EMB
#define DINKP  1344     // padded to 32
#define CATK   2048     // 2*DH
#define VOCAB  23262
#define VOCABP 23264    // padded to 16

typedef unsigned short u16;
typedef __attribute__((ext_vector_type(16))) __bf16 v16bf;
typedef __attribute__((ext_vector_type(8)))  float  v8f;

union BF16Tile { v16bf v; uint4 q[2]; };

__device__ __forceinline__ u16 f2bf(float f) {
  union { float f; unsigned int u; } x; x.f = f;
  unsigned int u = x.u;
  return (u16)((u + 0x7FFFu + ((u >> 16) & 1u)) >> 16);  // RNE
}
__device__ __forceinline__ float sigm(float x) { return 1.f / (1.f + __expf(-x)); }

// ---------------- packing / gather kernels ----------------

// W [N,K] f32 row-major  ->  dst [Np,Kp] bf16, zero padded
__global__ void pack_weight(const float* __restrict__ W, u16* __restrict__ dst,
                            int N, int K, int Np, int Kp) {
  size_t i = (size_t)blockIdx.x * blockDim.x + threadIdx.x;
  size_t total = (size_t)Np * Kp;
  if (i >= total) return;
  int n = (int)(i / Kp), k = (int)(i % Kp);
  dst[i] = (n < N && k < K) ? f2bf(W[(size_t)n * K + k]) : (u16)0;
}

// src f32 rows (row stride srcld) -> dst [rows,Kp] bf16 zero padded
__global__ void pack_act(const float* __restrict__ src, int srcld,
                         u16* __restrict__ dst, int K, int Kp, int total) {
  int i = blockIdx.x * blockDim.x + threadIdx.x;
  if (i >= total) return;
  int r = i / Kp, k = i % Kp;
  dst[i] = (k < K) ? f2bf(src[(size_t)r * srcld + k]) : (u16)0;
}

// embedding gather fused with bf16 pack: dst [rows, EMBP]
__global__ void gather_emb(const int* __restrict__ idx, const float* __restrict__ table,
                           u16* __restrict__ dst, int rows) {
  int i = blockIdx.x * blockDim.x + threadIdx.x;
  if (i >= rows * EMBP) return;
  int r = i / EMBP, e = i % EMBP;
  dst[i] = (e < EMB) ? f2bf(table[(size_t)idx[r] * EMB + e]) : (u16)0;
}

__global__ void zero_f32(float* __restrict__ p, size_t n) {
  size_t i = (size_t)blockIdx.x * blockDim.x + threadIdx.x;
  if (i < n) p[i] = 0.f;
}

// ---------------- WMMA GEMM:  C[M,N] = A[M,K] * W[N,K]^T (+bias / +=C / tanh) ----
// A: bf16 [M, lda] (lda = K padded to 32); W: bf16 [Np, ldw]; C: f32 [M, ldc]
// grid = (ceil(N/16), M/64), block = 32 (one wave -> 64x16 tile of C)
// Software-pipelined: k-step tiles are prefetched one iteration ahead so the
// four back-to-back WMMAs overlap with the next load group.
#define GF_ACC  1
#define GF_TANH 2

__global__ __launch_bounds__(32)
void gemm_wmma(const u16* __restrict__ A, int lda,
               const u16* __restrict__ W, int ldw,
               const float* __restrict__ bias,
               float* __restrict__ C, int ldc,
               int K, int N, int flags)
{
  const int lane = threadIdx.x & 31;
  const int r    = lane & 15;
  const int hi   = lane >> 4;
  const int n0   = blockIdx.x << 4;
  const int m0   = blockIdx.y << 6;

  v8f zero = {0.f,0.f,0.f,0.f,0.f,0.f,0.f,0.f};
  v8f acc[4] = {zero, zero, zero, zero};

  // B operand: lane = N column (n0+r), K chunks [8*hi .. +7] and [16+8*hi .. +7]
  const u16* wp = W + (size_t)(n0 + r) * ldw + (hi << 3);
  // A operand: lane = M row, K chunks [8*hi .. +7] and [16+8*hi .. +7]
  const u16* ap = A + (size_t)(m0 + r) * lda + (hi << 3);
  const size_t mstride = (size_t)16 * lda;

  // prologue: load tiles for k0 = 0
  BF16Tile bt, at[4];
  bt.q[0] = *(const uint4*)(wp);
  bt.q[1] = *(const uint4*)(wp + 16);
#pragma unroll
  for (int mt = 0; mt < 4; ++mt) {
    const u16* a0 = ap + mt * mstride;
    at[mt].q[0] = *(const uint4*)(a0);
    at[mt].q[1] = *(const uint4*)(a0 + 16);
  }

  for (int k0 = 32; k0 < K; k0 += 32) {
    // prefetch next k-step while current WMMAs run
    BF16Tile btn, atn[4];
    btn.q[0] = *(const uint4*)(wp + k0);
    btn.q[1] = *(const uint4*)(wp + k0 + 16);
#pragma unroll
    for (int mt = 0; mt < 4; ++mt) {
      const u16* a0 = ap + mt * mstride + k0;
      atn[mt].q[0] = *(const uint4*)(a0);
      atn[mt].q[1] = *(const uint4*)(a0 + 16);
    }
#pragma unroll
    for (int mt = 0; mt < 4; ++mt)
      acc[mt] = __builtin_amdgcn_wmma_f32_16x16x32_bf16(
          false, at[mt].v, false, bt.v, (short)0, acc[mt], false, false);
    bt = btn;
#pragma unroll
    for (int mt = 0; mt < 4; ++mt) at[mt] = atn[mt];
  }

  // drain last k-step
#pragma unroll
  for (int mt = 0; mt < 4; ++mt)
    acc[mt] = __builtin_amdgcn_wmma_f32_16x16x32_bf16(
        false, at[mt].v, false, bt.v, (short)0, acc[mt], false, false);

  const int n = n0 + r;
  if (n < N) {
    const float bv = bias ? bias[n] : 0.f;
#pragma unroll
    for (int mt = 0; mt < 4; ++mt) {
#pragma unroll
      for (int v = 0; v < 8; ++v) {
        int m = m0 + mt * 16 + (hi << 3) + v;   // C: VGPR v -> M = v + 8*hi
        float* cp = C + (size_t)m * ldc + n;
        float val = acc[mt][v] + bv;
        if (flags & GF_ACC)  val += *cp;
        if (flags & GF_TANH) val = tanhf(val);
        *cp = val;
      }
    }
  }
}

// ---------------- elementwise / attention kernels ----------------

// gates g[B,4H] (i,f,g,o) ; c in-place ; h -> hout with row stride
__global__ void lstm_cell(const float* __restrict__ g, float* __restrict__ c,
                          float* __restrict__ hout, int hstride, int H) {
  int i = blockIdx.x * blockDim.x + threadIdx.x;
  if (i >= BATCH * H) return;
  int b = i / H, j = i % H;
  const float* gr = g + (size_t)b * 4 * H;
  float ig = sigm(gr[j]);
  float fg = sigm(gr[H + j]);
  float gg = tanhf(gr[2 * H + j]);
  float og = sigm(gr[3 * H + j]);
  float c2 = fg * c[i] + ig * gg;
  c[i] = c2;
  hout[(size_t)b * hstride + j] = og * tanhf(c2);
}

// out[b, 2j]=a[b,j], out[b, 2j+1]=bb[b,j]  (stack(...,-1).reshape)
__global__ void interleave2(const float* __restrict__ a, int sa,
                            const float* __restrict__ bb, int sb,
                            float* __restrict__ out) {
  int i = blockIdx.x * blockDim.x + threadIdx.x;
  if (i >= BATCH * EH) return;
  int b = i / EH, j = i % EH;
  out[(size_t)b * DH + 2 * j]     = a[(size_t)b * sa + j];
  out[(size_t)b * DH + 2 * j + 1] = bb[(size_t)b * sb + j];
}

// score[s,b] = dot(enc[s,b,:], semi[b,:]) ; one wave per (s,b)
__global__ __launch_bounds__(128)
void attn_score(const float* __restrict__ enc, const float* __restrict__ semi,
                float* __restrict__ score) {
  int wave = (blockIdx.x * blockDim.x + threadIdx.x) >> 5;
  int lane = threadIdx.x & 31;
  if (wave >= S_LEN * BATCH) return;
  int s = wave / BATCH, b = wave % BATCH;
  const float* e = enc + ((size_t)s * BATCH + b) * DH;
  const float* m = semi + (size_t)b * DH;
  float acc = 0.f;
  for (int d = lane; d < DH; d += 32) acc += e[d] * m[d];
  for (int off = 16; off; off >>= 1) acc += __shfl_xor(acc, off, 32);
  if (lane == 0) score[wave] = acc;
}

__global__ void softmax_s(const float* __restrict__ score, float* __restrict__ alpha) {
  int b = blockIdx.x * blockDim.x + threadIdx.x;
  if (b >= BATCH) return;
  float mx = -1e30f;
  for (int s = 0; s < S_LEN; ++s) mx = fmaxf(mx, score[s * BATCH + b]);
  float sum = 0.f;
  for (int s = 0; s < S_LEN; ++s) sum += __expf(score[s * BATCH + b] - mx);
  float inv = 1.f / sum;
  for (int s = 0; s < S_LEN; ++s)
    alpha[s * BATCH + b] = __expf(score[s * BATCH + b] - mx) * inv;
}

// st[b,d] = sum_s alpha[s,b] * enc[s,b,d]
__global__ void calc_stilde(const float* __restrict__ alpha, const float* __restrict__ enc,
                            float* __restrict__ st) {
  int i = blockIdx.x * blockDim.x + threadIdx.x;
  if (i >= BATCH * DH) return;
  int b = i / DH, d = i % DH;
  float acc = 0.f;
  for (int s = 0; s < S_LEN; ++s)
    acc += alpha[s * BATCH + b] * enc[((size_t)s * BATCH + b) * DH + d];
  st[i] = acc;
}

// cat([s_tilde, h]) -> bf16 [B, CATK]
__global__ void cat_st_h(const float* __restrict__ st, const float* __restrict__ h,
                         u16* __restrict__ dst) {
  int i = blockIdx.x * blockDim.x + threadIdx.x;
  if (i >= BATCH * CATK) return;
  int b = i / CATK, k = i % CATK;
  float v = (k < DH) ? st[(size_t)b * DH + k] : h[(size_t)b * DH + (k - DH)];
  dst[i] = f2bf(v);
}

// cat([c_t, trg_emb]) -> bf16 [B, DINKP]  (trg already bf16-packed rows of EMBP)
__global__ void build_dec_in(const float* __restrict__ ct, const u16* __restrict__ trg,
                             u16* __restrict__ dst) {
  int i = blockIdx.x * blockDim.x + threadIdx.x;
  if (i >= BATCH * DINKP) return;
  int b = i / DINKP, k = i % DINKP;
  u16 v;
  if (k < DH)        v = f2bf(ct[(size_t)b * DH + k]);
  else if (k < DINK) v = trg[(size_t)b * EMBP + (k - DH)];
  else               v = 0;
  dst[i] = v;
}

// in-place log_softmax over VOCAB per row; one block per row
__global__ __launch_bounds__(256)
void log_softmax_row(float* __restrict__ x) {
  __shared__ float red[256];
  float* row = x + (size_t)blockIdx.x * VOCAB;
  int t = threadIdx.x;
  float mx = -1e30f;
  for (int v = t; v < VOCAB; v += 256) mx = fmaxf(mx, row[v]);
  red[t] = mx; __syncthreads();
  for (int o = 128; o; o >>= 1) { if (t < o) red[t] = fmaxf(red[t], red[t + o]); __syncthreads(); }
  mx = red[0]; __syncthreads();
  float sum = 0.f;
  for (int v = t; v < VOCAB; v += 256) sum += __expf(row[v] - mx);
  red[t] = sum; __syncthreads();
  for (int o = 128; o; o >>= 1) { if (t < o) red[t] += red[t + o]; __syncthreads(); }
  float lse = mx + __logf(red[0]);
  for (int v = t; v < VOCAB; v += 256) row[v] -= lse;
}

// ---------------- host side ----------------

static inline void launch_gemm(hipStream_t st, const u16* A, int lda, const u16* W, int ldw,
                               const float* bias, float* C, int ldc,
                               int M, int K, int N, int flags) {
  dim3 grid((unsigned)((N + 15) / 16), (unsigned)(M / 64));
  gemm_wmma<<<grid, 32, 0, st>>>(A, lda, W, ldw, bias, C, ldc, K, N, flags);
}

static inline void launch_packW(hipStream_t s, const float* src, u16* dst,
                                int N, int K, int Np, int Kp) {
  size_t tot = (size_t)Np * Kp;
  pack_weight<<<(unsigned)((tot + 255) / 256), 256, 0, s>>>(src, dst, N, K, Np, Kp);
}

extern "C" void kernel_launch(void* const* d_in, const int* in_sizes, int n_in,
                              void* d_out, int out_size, void* d_ws, size_t ws_size,
                              hipStream_t stream) {
  (void)in_sizes; (void)n_in; (void)out_size; (void)ws_size;

  const int*   src_idx = (const int*)d_in[0];
  const int*   trg_idx = (const int*)d_in[1];
  const float* EEMBw   = (const float*)d_in[2];
  const float* eWihF   = (const float*)d_in[3];
  const float* eWhhF   = (const float*)d_in[4];
  const float* ebF     = (const float*)d_in[5];
  const float* eWihR   = (const float*)d_in[6];
  const float* eWhhR   = (const float*)d_in[7];
  const float* ebR     = (const float*)d_in[8];
  const float* aWi     = (const float*)d_in[9];
  const float* aWo     = (const float*)d_in[10];
  const float* dWih    = (const float*)d_in[11];
  const float* dWhh    = (const float*)d_in[12];
  const float* db      = (const float*)d_in[13];
  const float* DEMBw   = (const float*)d_in[14];
  const float* gW      = (const float*)d_in[15];
  const float* gb      = (const float*)d_in[16];
  float* out = (float*)d_out;

  // ---- workspace carve-out (256B aligned) ----
  char* p = (char*)d_ws;
  auto alloc = [&](size_t bytes) -> char* {
    char* r = p; p += (bytes + 255) & ~(size_t)255; return r;
  };
  u16* wEihF = (u16*)alloc(sizeof(u16) * (size_t)G_EH * EMBP);
  u16* wEhhF = (u16*)alloc(sizeof(u16) * (size_t)G_EH * EH);
  u16* wEihR = (u16*)alloc(sizeof(u16) * (size_t)G_EH * EMBP);
  u16* wEhhR = (u16*)alloc(sizeof(u16) * (size_t)G_EH * EH);
  u16* wAWi  = (u16*)alloc(sizeof(u16) * (size_t)DH * DH);
  u16* wAWo  = (u16*)alloc(sizeof(u16) * (size_t)DH * CATK);
  u16* wDWih = (u16*)alloc(sizeof(u16) * (size_t)G_DH * DINKP);
  u16* wDWhh = (u16*)alloc(sizeof(u16) * (size_t)G_DH * DH);
  u16* wGen  = (u16*)alloc(sizeof(u16) * (size_t)VOCABP * DH);
  u16* embS  = (u16*)alloc(sizeof(u16) * (size_t)S_LEN * BATCH * EMBP);
  u16* embT  = (u16*)alloc(sizeof(u16) * (size_t)(T_LEN - 1) * BATCH * EMBP);
  u16* hpack = (u16*)alloc(sizeof(u16) * (size_t)BATCH * DH);
  u16* cat1p = (u16*)alloc(sizeof(u16) * (size_t)BATCH * CATK);
  u16* dinp  = (u16*)alloc(sizeof(u16) * (size_t)BATCH * DINKP);
  float* enc   = (float*)alloc(sizeof(float) * (size_t)S_LEN * BATCH * DH); // [s,b,1024]: f|r
  float* g_buf = (float*)alloc(sizeof(float) * (size_t)BATCH * G_DH);       // gates (max 4096)
  float* cF    = (float*)alloc(sizeof(float) * (size_t)BATCH * EH);
  float* cR    = (float*)alloc(sizeof(float) * (size_t)BATCH * EH);
  float* h_dec = (float*)alloc(sizeof(float) * (size_t)BATCH * DH);
  float* c_dec = (float*)alloc(sizeof(float) * (size_t)BATCH * DH);
  float* semi  = (float*)alloc(sizeof(float) * (size_t)BATCH * DH);
  float* score = (float*)alloc(sizeof(float) * (size_t)S_LEN * BATCH);
  float* alpha = (float*)alloc(sizeof(float) * (size_t)S_LEN * BATCH);
  float* stil  = (float*)alloc(sizeof(float) * (size_t)BATCH * DH);
  float* c_t   = (float*)alloc(sizeof(float) * (size_t)BATCH * DH);

  // ---- one-time weight bf16 packing (L2-resident thereafter) ----
  launch_packW(stream, eWihF, wEihF, G_EH, EMB, G_EH, EMBP);
  launch_packW(stream, eWhhF, wEhhF, G_EH, EH, G_EH, EH);
  launch_packW(stream, eWihR, wEihR, G_EH, EMB, G_EH, EMBP);
  launch_packW(stream, eWhhR, wEhhR, G_EH, EH, G_EH, EH);
  launch_packW(stream, aWi,   wAWi,  DH, DH, DH, DH);
  launch_packW(stream, aWo,   wAWo,  DH, CATK, DH, CATK);
  launch_packW(stream, dWih,  wDWih, G_DH, DINK, G_DH, DINKP);
  launch_packW(stream, dWhh,  wDWhh, G_DH, DH, G_DH, DH);
  launch_packW(stream, gW,    wGen,  VOCAB, DH, VOCABP, DH);

  // ---- embedding gathers (fused f32->bf16) ----
  gather_emb<<<(S_LEN * BATCH * EMBP + 255) / 256, 256, 0, stream>>>(src_idx, EEMBw, embS, S_LEN * BATCH);
  gather_emb<<<((T_LEN - 1) * BATCH * EMBP + 255) / 256, 256, 0, stream>>>(trg_idx, DEMBw, embT, (T_LEN - 1) * BATCH);

  // ---- init states / output slab 0 ----
  zero_f32<<<(BATCH * EH + 255) / 256, 256, 0, stream>>>(cF, (size_t)BATCH * EH);
  zero_f32<<<(BATCH * EH + 255) / 256, 256, 0, stream>>>(cR, (size_t)BATCH * EH);
  {
    size_t n0 = (size_t)BATCH * VOCAB;
    zero_f32<<<(unsigned)((n0 + 255) / 256), 256, 0, stream>>>(out, n0);
  }

  const int cellEH = (BATCH * EH + 255) / 256;
  const int cellDH = (BATCH * DH + 255) / 256;

  // ---- encoder forward scan ----
  for (int t = 0; t < S_LEN; ++t) {
    launch_gemm(stream, embS + (size_t)t * BATCH * EMBP, EMBP, wEihF, EMBP,
                ebF, g_buf, G_EH, BATCH, EMB, G_EH, 0);
    if (t > 0) {
      pack_act<<<cellEH, 256, 0, stream>>>(enc + (size_t)(t - 1) * BATCH * DH, DH,
                                           hpack, EH, EH, BATCH * EH);
      launch_gemm(stream, hpack, EH, wEhhF, EH, nullptr, g_buf, G_EH, BATCH, EH, G_EH, GF_ACC);
    }
    lstm_cell<<<cellEH, 256, 0, stream>>>(g_buf, cF, enc + (size_t)t * BATCH * DH, DH, EH);
  }

  // ---- encoder reverse scan (writes enc[.., 512:1024]) ----
  for (int t = S_LEN - 1; t >= 0; --t) {
    launch_gemm(stream, embS + (size_t)t * BATCH * EMBP, EMBP, wEihR, EMBP,
                ebR, g_buf, G_EH, BATCH, EMB, G_EH, 0);
    if (t < S_LEN - 1) {
      pack_act<<<cellEH, 256, 0, stream>>>(enc + (size_t)(t + 1) * BATCH * DH + EH, DH,
                                           hpack, EH, EH, BATCH * EH);
      launch_gemm(stream, hpack, EH, wEhhR, EH, nullptr, g_buf, G_EH, BATCH, EH, G_EH, GF_ACC);
    }
    lstm_cell<<<cellEH, 256, 0, stream>>>(g_buf, cR, enc + (size_t)t * BATCH * DH + EH, DH, EH);
  }

  // ---- decoder initial state: interleave (hF,hR) and (cF,cR) ----
  interleave2<<<cellEH, 256, 0, stream>>>(enc + (size_t)(S_LEN - 1) * BATCH * DH, DH,
                                          enc + EH, DH, h_dec);
  interleave2<<<cellEH, 256, 0, stream>>>(cF, EH, cR, EH, c_dec);

  // hpack <- h_dec (bf16)
  pack_act<<<cellDH, 256, 0, stream>>>(h_dec, DH, hpack, DH, DH, BATCH * DH);

  // ---- decoder scan ----
  for (int t = 0; t < T_LEN - 1; ++t) {
    // attention
    launch_gemm(stream, hpack, DH, wAWi, DH, nullptr, semi, DH, BATCH, DH, DH, 0);
    attn_score<<<(S_LEN * BATCH * 32 + 127) / 128, 128, 0, stream>>>(enc, semi, score);
    softmax_s<<<1, BATCH, 0, stream>>>(score, alpha);
    calc_stilde<<<cellDH, 256, 0, stream>>>(alpha, enc, stil);
    cat_st_h<<<(BATCH * CATK + 255) / 256, 256, 0, stream>>>(stil, h_dec, cat1p);
    launch_gemm(stream, cat1p, CATK, wAWo, CATK, nullptr, c_t, DH, BATCH, CATK, DH, GF_TANH);
    // decoder LSTM
    build_dec_in<<<(BATCH * DINKP + 255) / 256, 256, 0, stream>>>(
        c_t, embT + (size_t)t * BATCH * EMBP, dinp);
    launch_gemm(stream, dinp, DINKP, wDWih, DINKP, db, g_buf, G_DH, BATCH, DINK, G_DH, 0);
    launch_gemm(stream, hpack, DH, wDWhh, DH, nullptr, g_buf, G_DH, BATCH, DH, G_DH, GF_ACC);
    lstm_cell<<<cellDH, 256, 0, stream>>>(g_buf, c_dec, h_dec, DH, DH);
    pack_act<<<cellDH, 256, 0, stream>>>(h_dec, DH, hpack, DH, DH, BATCH * DH);
    // generator + log-softmax straight into output slab t+1
    float* slab = out + (size_t)(t + 1) * BATCH * VOCAB;
    launch_gemm(stream, hpack, DH, wGen, DH, gb, slab, VOCAB, BATCH, DH, VOCAB, 0);
    log_softmax_row<<<BATCH, 256, 0, stream>>>(slab);
  }
}
